// GraphAttentionLayer_74887049773681
// MI455X (gfx1250) — matrix operand order repaired and checked by reading, not compiled, not run on hardware
//
#include <hip/hip_runtime.h>
#include <math.h>

// Problem constants (from reference)
#define Bc  2
#define Nc  10000
#define Ec  160000
#define Hc  128
#define ETc 8
#define BNc (Bc*Nc)          // 20000 rows, flattened (b,n)
#define BEc (Bc*Ec)          // 320000 (b,e) pairs
#define MTILES (BNc/16)      // 1250 16-row tiles (exact)

typedef __attribute__((ext_vector_type(16))) _Float16 v16h;
typedef __attribute__((ext_vector_type(8)))  _Float16 v8h;
typedef __attribute__((ext_vector_type(4)))  _Float16 v4h;
typedef __attribute__((ext_vector_type(8)))  float    v8f;

// ---------------------------------------------------------------------------
// WMMA helpers (CDNA5 16x16x32 f16 -> f32)
// ---------------------------------------------------------------------------
__device__ __forceinline__ v8f wmma16(v16h a, v16h b, v8f c) {
  // (neg_a, A, neg_b, B, c_mod, C, reuse_a, reuse_b)
  return __builtin_amdgcn_wmma_f32_16x16x32_f16(false, a, false, b, (short)0, c,
                                                false, false);
}

// A-fragment from a row-major f16 LDS tile [16][ldk].
// ISA 16-bit A 16x32 layout: lanes 0-15 -> M=0..15, halves e<8: K=e, e>=8: K=16+(e-8);
// lanes 16-31 shift K by +8 in each group.
__device__ __forceinline__ v16h load_a_frag(const _Float16* t, int ldk, int kt, int lane) {
  int m = lane & 15, hi = lane >> 4;
  const _Float16* row = t + m * ldk + kt * 32 + hi * 8;
  v8h lo = *(const v8h*)(row);
  v8h hh = *(const v8h*)(row + 16);
  v16h a;
#pragma unroll
  for (int i = 0; i < 8; ++i) { a[i] = lo[i]; a[i + 8] = hh[i]; }
  return a;
}

// K-loop over kt tiles, 8 N-tiles (128 output cols), B from fragment-ordered buffer:
// frag base = ((kt*8+nt)*32 + lane)*16 halves (32B contiguous per lane).
__device__ __forceinline__ void gemm_tile(const _Float16* lds_tile, int ldk, int kts,
                                          const _Float16* __restrict__ WF,
                                          v8f acc[8], int lane) {
  for (int kt = 0; kt < kts; ++kt) {
    v16h a = load_a_frag(lds_tile, ldk, kt, lane);
#pragma unroll
    for (int nt = 0; nt < 8; ++nt) {
      v16h b = *(const v16h*)(WF + (size_t)(((kt << 3) + nt) * 32 + lane) * 16);
      acc[nt] = wmma16(a, b, acc[nt]);
    }
  }
}

// ---------------------------------------------------------------------------
// Monotone float<->uint encoding for atomic segment-max
// ---------------------------------------------------------------------------
__device__ __forceinline__ unsigned fenc(float x) {
  unsigned u = __float_as_uint(x);
  return (u >> 31) ? ~u : (u | 0x80000000u);
}
__device__ __forceinline__ float fdec(unsigned e) {
  unsigned u = (e >> 31) ? (e & 0x7FFFFFFFu) : ~e;   // fdec(0) = NaN (empty segment)
  return __uint_as_float(u);
}

// ---------------------------------------------------------------------------
// Weight -> f16 B-fragment conversion. Dense B 32x16 layout:
// lanes 0-15 hold K=kt*32+{0..7,16..23} (halves e=0..15), lanes 16-31 hold K=kt*32+{8..15,24..31}.
// ---------------------------------------------------------------------------
__global__ void prep_frag_kernel(const float* __restrict__ W, _Float16* __restrict__ dst,
                                 int Krows) {
  int id = blockIdx.x * blockDim.x + threadIdx.x;
  if (id >= Krows * Hc) return;
  int e    = id & 15;
  int lane = (id >> 4) & 31;
  int fi   = id >> 9;          // frag index = kt*8 + nt
  int nt   = fi & 7;
  int kt   = fi >> 3;
  int kk   = kt * 32 + ((e >> 3) << 4) + ((lane >> 4) << 3) + (e & 7);
  int nn   = nt * 16 + (lane & 15);
  dst[id]  = (_Float16)W[(size_t)kk * Hc + nn];
}

// Per-edge-type projections: Ktype = emb@Wk, Vtype = emb@Wv + bv, ebias = emb@We + be
__global__ void prep_type_kernel(const float* __restrict__ edge_emb,
                                 const float* __restrict__ Wk, const float* __restrict__ Wv,
                                 const float* __restrict__ bv, const float* __restrict__ We,
                                 const float* __restrict__ be,
                                 float* __restrict__ ktype, float* __restrict__ vtype,
                                 float* __restrict__ ebias) {
  int t = blockIdx.x, n = threadIdx.x;
  float sk = 0.f, sv = 0.f;
  for (int k = 0; k < Hc; ++k) {
    float ev = edge_emb[t * Hc + k];
    sk += ev * Wk[k * Hc + n];
    sv += ev * Wv[k * Hc + n];
  }
  ktype[t * Hc + n] = sk;
  vtype[t * Hc + n] = sv + bv[n];
  if (n == 0) {
    float s = 0.f;
    for (int k = 0; k < Hc; ++k) s += edge_emb[t * Hc + k] * We[k];
    ebias[t] = s + be[0];
  }
}

__global__ void init_kernel(float* __restrict__ agg, float* __restrict__ segsum,
                            unsigned* __restrict__ segmax) {
  int i = blockIdx.x * blockDim.x + threadIdx.x;
  if (i < BNc * Hc) agg[i] = 0.f;
  if (i < BNc) { segsum[i] = 0.f; segmax[i] = 0u; }
}

// ---------------------------------------------------------------------------
// Node projections: Q = (h+t)@Wq+bq, Kn = (h+t)@Wk+bk, Vn = h@Wv   (WMMA)
// ---------------------------------------------------------------------------
__device__ __forceinline__ void store_tile(float* __restrict__ out, const v8f acc[8],
                                           const float* __restrict__ bias,
                                           int tile, int lane) {
  int col0 = lane & 15, hi = lane >> 4;
  int g0 = tile * 16;
#pragma unroll
  for (int nt = 0; nt < 8; ++nt) {
    int col = nt * 16 + col0;
    float bb = bias ? bias[col] : 0.f;
#pragma unroll
    for (int r = 0; r < 8; ++r)
      out[(size_t)(g0 + r + hi * 8) * Hc + col] = acc[nt][r] + bb;
  }
}

__global__ void qkv_kernel(const float* __restrict__ hidden, const float* __restrict__ time_emb,
                           const _Float16* __restrict__ WqF, const _Float16* __restrict__ WkF,
                           const _Float16* __restrict__ WvF,
                           const float* __restrict__ bq, const float* __restrict__ bk,
                           float* __restrict__ Q, float* __restrict__ Kn, float* __restrict__ Vn) {
  __shared__ alignas(16) _Float16 smem[4][2][16 * Hc];   // 32 KB: XA=(h+t), XH=h per wave
  int wave = threadIdx.x >> 5, lane = threadIdx.x & 31;
  int tile = blockIdx.x * 4 + wave;
  bool active = tile < MTILES;
  _Float16* XA = smem[wave][0];
  _Float16* XH = smem[wave][1];
  if (active) {
    int g0 = tile * 16;
#pragma unroll
    for (int it = 0; it < 16; ++it) {
      int pos = it * 32 + lane;           // 512 float4 per tile
      int row = pos >> 5, c4 = pos & 31;
      const float4 h4 = *(const float4*)(hidden + (size_t)(g0 + row) * Hc + c4 * 4);
      const float4 t4 = *(const float4*)(time_emb + (size_t)(g0 + row) * Hc + c4 * 4);
      v4h xa, xh;
      xa.x = (_Float16)(h4.x + t4.x); xa.y = (_Float16)(h4.y + t4.y);
      xa.z = (_Float16)(h4.z + t4.z); xa.w = (_Float16)(h4.w + t4.w);
      xh.x = (_Float16)h4.x; xh.y = (_Float16)h4.y;
      xh.z = (_Float16)h4.z; xh.w = (_Float16)h4.w;
      *(v4h*)(XA + row * Hc + c4 * 4) = xa;
      *(v4h*)(XH + row * Hc + c4 * 4) = xh;
    }
  }
  __syncthreads();
  if (!active) return;
  {
    v8f acc[8] = {};
    gemm_tile(XA, Hc, 4, WqF, acc, lane);
    store_tile(Q, acc, bq, tile, lane);
  }
  {
    v8f acc[8] = {};
    gemm_tile(XA, Hc, 4, WkF, acc, lane);
    store_tile(Kn, acc, bk, tile, lane);
  }
  {
    v8f acc[8] = {};
    gemm_tile(XH, Hc, 4, WvF, acc, lane);
    store_tile(Vn, acc, nullptr, tile, lane);
  }
}

// ---------------------------------------------------------------------------
// Edge phase: one wave per (b,e).  logit = q . (Knode[src]+Ktype[t]) / sqrt(H) + ebias
// ---------------------------------------------------------------------------
__global__ void edge_logits_kernel(const int* __restrict__ eidx, const int* __restrict__ etype,
                                   const float* __restrict__ Q, const float* __restrict__ Kn,
                                   const float* __restrict__ ktype, const float* __restrict__ ebias,
                                   float* __restrict__ logits, unsigned* __restrict__ segmax) {
  int wid  = (blockIdx.x * blockDim.x + threadIdx.x) >> 5;
  int lane = threadIdx.x & 31;
  if (wid >= BEc) return;
  int b = wid / Ec, e = wid - b * Ec;
  int src = eidx[e], tgt = eidx[Ec + e], t = etype[e];
  const float4 q  = ((const float4*)(Q     + (size_t)(b * Nc + tgt) * Hc))[lane];
  const float4 k  = ((const float4*)(Kn    + (size_t)(b * Nc + src) * Hc))[lane];
  const float4 kt = ((const float4*)(ktype + (size_t)t * Hc))[lane];
  float p = q.x * (k.x + kt.x) + q.y * (k.y + kt.y) +
            q.z * (k.z + kt.z) + q.w * (k.w + kt.w);
#pragma unroll
  for (int off = 16; off; off >>= 1) p += __shfl_xor(p, off, 32);
  if (lane == 0) {
    float logit = p * 0.08838834764831845f + ebias[t];   // 1/sqrt(128)
    logits[wid] = logit;
    atomicMax(&segmax[b * Nc + tgt], fenc(logit));
  }
}

__global__ void exp_sum_kernel(const int* __restrict__ eidx,
                               const float* __restrict__ logits,
                               const unsigned* __restrict__ segmax,
                               float* __restrict__ exw, float* __restrict__ segsum) {
  int i = blockIdx.x * blockDim.x + threadIdx.x;
  if (i >= BEc) return;
  int b = i / Ec, e = i - b * Ec;
  int tgt = eidx[Ec + e];
  float m = fdec(segmax[b * Nc + tgt]);
  if ((__float_as_uint(m) & 0x7F800000u) == 0x7F800000u) m = 0.f;  // !isfinite -> 0
  float ex = __expf(logits[i] - m);
  exw[i] = ex;
  atomicAdd(&segsum[b * Nc + tgt], ex);
}

__global__ void scatter_kernel(const int* __restrict__ eidx, const int* __restrict__ etype,
                               const float* __restrict__ Vn, const float* __restrict__ vtype,
                               const float* __restrict__ exw, const float* __restrict__ segsum,
                               float* __restrict__ agg) {
  int wid  = (blockIdx.x * blockDim.x + threadIdx.x) >> 5;
  int lane = threadIdx.x & 31;
  if (wid >= BEc) return;
  int b = wid / Ec, e = wid - b * Ec;
  int src = eidx[e], tgt = eidx[Ec + e], t = etype[e];
  float w = exw[wid] / segsum[b * Nc + tgt];
  const float4 v  = ((const float4*)(Vn    + (size_t)(b * Nc + src) * Hc))[lane];
  const float4 vt = ((const float4*)(vtype + (size_t)t * Hc))[lane];
  float* dst = agg + (size_t)(b * Nc + tgt) * Hc + lane * 4;
  atomicAdd(dst + 0, (v.x + vt.x) * w);
  atomicAdd(dst + 1, (v.y + vt.y) * w);
  atomicAdd(dst + 2, (v.z + vt.z) * w);
  atomicAdd(dst + 3, (v.w + vt.w) * w);
}

// ---------------------------------------------------------------------------
// FFN + residual + LayerNorm, fused per 16-row tile (2 WMMA GEMMs, K=384 then K=128)
// ---------------------------------------------------------------------------
__global__ void ffn_ln_kernel(const float* __restrict__ hidden, const float* __restrict__ agg,
                              const float* __restrict__ time_emb,
                              const _Float16* __restrict__ W1F, const _Float16* __restrict__ W2F,
                              const float* __restrict__ b1, const float* __restrict__ b2,
                              const float* __restrict__ gamma, const float* __restrict__ beta,
                              float* __restrict__ out) {
  __shared__ alignas(16) _Float16 smem[2][16 * 384 + 16 * Hc];   // 32 KB (2 waves)
  int wave = threadIdx.x >> 5, lane = threadIdx.x & 31;
  int tile = blockIdx.x * 2 + wave;
  bool active = tile < MTILES;
  _Float16* TA = smem[wave];
  _Float16* TH = smem[wave] + 16 * 384;
  if (active) {
    int g0 = tile * 16;
    const float* srcs[3] = {hidden, agg, time_emb};
#pragma unroll
    for (int s = 0; s < 3; ++s) {
      const float* S = srcs[s];
#pragma unroll
      for (int it = 0; it < 16; ++it) {
        int pos = it * 32 + lane;
        int row = pos >> 5, c4 = pos & 31;
        float4 v = *(const float4*)(S + (size_t)(g0 + row) * Hc + c4 * 4);
        v4h h;
        h.x = (_Float16)v.x; h.y = (_Float16)v.y;
        h.z = (_Float16)v.z; h.w = (_Float16)v.w;
        *(v4h*)(TA + row * 384 + s * Hc + c4 * 4) = h;
      }
    }
  }
  __syncthreads();
  int col0 = lane & 15, hi = lane >> 4;
  if (active) {
    v8f acc[8] = {};
    gemm_tile(TA, 384, 12, W1F, acc, lane);
    // silu epilogue -> TH (f16), C/D layout: row = r + 8*hi, col = nt*16 + col0
#pragma unroll
    for (int nt = 0; nt < 8; ++nt) {
      int col = nt * 16 + col0;
      float bb = b1[col];
#pragma unroll
      for (int r = 0; r < 8; ++r) {
        float x = acc[nt][r] + bb;
        TH[(r + hi * 8) * Hc + col] = (_Float16)(x / (1.f + __expf(-x)));
      }
    }
  }
  __syncthreads();
  if (!active) return;
  v8f acc2[8] = {};
  gemm_tile(TH, Hc, 4, W2F, acc2, lane);
  int g0 = tile * 16;
#pragma unroll
  for (int r = 0; r < 8; ++r) {
    int row = g0 + r + hi * 8;
    float vals[8];
    float sum = 0.f, sq = 0.f;
#pragma unroll
    for (int nt = 0; nt < 8; ++nt) {
      int col = nt * 16 + col0;
      float x = hidden[(size_t)row * Hc + col] + acc2[nt][r] + b2[col];
      vals[nt] = x; sum += x; sq += x * x;
    }
    // row lives in one 16-lane half: width-16 butterfly reduction
#pragma unroll
    for (int off = 8; off; off >>= 1) {
      sum += __shfl_xor(sum, off, 32);
      sq  += __shfl_xor(sq,  off, 32);
    }
    float mean = sum * (1.f / 128.f);
    float var  = sq * (1.f / 128.f) - mean * mean;
    float rstd = rsqrtf(var + 1e-5f);
#pragma unroll
    for (int nt = 0; nt < 8; ++nt) {
      int col = nt * 16 + col0;
      out[(size_t)row * Hc + col] = (vals[nt] - mean) * rstd * gamma[col] + beta[col];
    }
  }
}

// ---------------------------------------------------------------------------
// Host launcher
// ---------------------------------------------------------------------------
extern "C" void kernel_launch(void* const* d_in, const int* in_sizes, int n_in,
                              void* d_out, int out_size, void* d_ws, size_t ws_size,
                              hipStream_t stream) {
  const float* hidden   = (const float*)d_in[0];
  const float* time_emb = (const float*)d_in[1];
  const int*   eidx     = (const int*)d_in[2];
  const int*   etype    = (const int*)d_in[3];
  const float* edge_emb = (const float*)d_in[4];
  const float* Wq = (const float*)d_in[5];  const float* bq = (const float*)d_in[6];
  const float* Wk = (const float*)d_in[7];  const float* bk = (const float*)d_in[8];
  const float* Wv = (const float*)d_in[9];  const float* bv = (const float*)d_in[10];
  const float* We = (const float*)d_in[11]; const float* be = (const float*)d_in[12];
  const float* W1 = (const float*)d_in[13]; const float* b1 = (const float*)d_in[14];
  const float* W2 = (const float*)d_in[15]; const float* b2 = (const float*)d_in[16];
  const float* gamma = (const float*)d_in[17];
  const float* beta  = (const float*)d_in[18];
  float* out = (float*)d_out;

  // Workspace partition (256B aligned slices)
  char* ws = (char*)d_ws;
  size_t off = 0;
  auto take = [&](size_t bytes) -> char* {
    char* p = ws + off;
    off += (bytes + 255) & ~(size_t)255;
    return p;
  };
  _Float16* WqF = (_Float16*)take((size_t)Hc * Hc * 2);
  _Float16* WkF = (_Float16*)take((size_t)Hc * Hc * 2);
  _Float16* WvF = (_Float16*)take((size_t)Hc * Hc * 2);
  _Float16* W1F = (_Float16*)take((size_t)3 * Hc * Hc * 2);
  _Float16* W2F = (_Float16*)take((size_t)Hc * Hc * 2);
  float* ktype  = (float*)take((size_t)ETc * Hc * 4);
  float* vtype  = (float*)take((size_t)ETc * Hc * 4);
  float* ebias  = (float*)take((size_t)ETc * 4);
  float* Q      = (float*)take((size_t)BNc * Hc * 4);
  float* Kn     = (float*)take((size_t)BNc * Hc * 4);
  float* Vn     = (float*)take((size_t)BNc * Hc * 4);
  float* logits = (float*)take((size_t)BEc * 4);
  float* exw    = (float*)take((size_t)BEc * 4);
  unsigned* segmax = (unsigned*)take((size_t)BNc * 4);
  float* segsum = (float*)take((size_t)BNc * 4);
  float* agg    = (float*)take((size_t)BNc * Hc * 4);

  // 1) weights -> f16 B-fragment order
  prep_frag_kernel<<<(Hc * Hc + 255) / 256, 256, 0, stream>>>(Wq, WqF, Hc);
  prep_frag_kernel<<<(Hc * Hc + 255) / 256, 256, 0, stream>>>(Wk, WkF, Hc);
  prep_frag_kernel<<<(Hc * Hc + 255) / 256, 256, 0, stream>>>(Wv, WvF, Hc);
  prep_frag_kernel<<<(3 * Hc * Hc + 255) / 256, 256, 0, stream>>>(W1, W1F, 3 * Hc);
  prep_frag_kernel<<<(Hc * Hc + 255) / 256, 256, 0, stream>>>(W2, W2F, Hc);
  // 2) per-edge-type projections
  prep_type_kernel<<<ETc, Hc, 0, stream>>>(edge_emb, Wk, Wv, bv, We, be, ktype, vtype, ebias);
  // 3) zero accumulators
  init_kernel<<<(BNc * Hc + 255) / 256, 256, 0, stream>>>(agg, segsum, segmax);
  // 4) node Q/K/V (WMMA)
  qkv_kernel<<<(MTILES + 3) / 4, 128, 0, stream>>>(hidden, time_emb, WqF, WkF, WvF,
                                                   bq, bk, Q, Kn, Vn);
  // 5) logits + segment max
  edge_logits_kernel<<<(BEc * 32 + 255) / 256, 256, 0, stream>>>(eidx, etype, Q, Kn,
                                                                 ktype, ebias, logits, segmax);
  // 6) exp + segment sum
  exp_sum_kernel<<<(BEc + 255) / 256, 256, 0, stream>>>(eidx, logits, segmax, exw, segsum);
  // 7) weighted message scatter
  scatter_kernel<<<(BEc * 32 + 255) / 256, 256, 0, stream>>>(eidx, etype, Vn, vtype,
                                                             exw, segsum, agg);
  // 8) FFN + residual + LayerNorm (WMMA)
  ffn_ln_kernel<<<(MTILES + 1) / 2, 64, 0, stream>>>(hidden, agg, time_emb, W1F, W2F,
                                                     b1, b2, gamma, beta, out);
  (void)in_sizes; (void)n_in; (void)out_size; (void)ws_size;
}
